// CausalSelfAttention_15410342658807
// MI455X (gfx1250) — matrix operand order repaired and checked by reading, not compiled, not run on hardware
//
#include <hip/hip_runtime.h>
#include <hip/hip_bf16.h>
#include <math.h>

// ---------------------------------------------------------------------------
// CDNA5 (gfx1250, wave32) implementation of the JL-block-sparse attention.
// Heavy matmuls: v_wmma_f32_16x16x32_bf16.
// GEMM staging: double-buffered GLOBAL_LOAD_ASYNC_TO_LDS (A tile) + Tensor
// Data Mover descriptor load (B tile), overlapped with WMMA compute.
// ---------------------------------------------------------------------------

typedef unsigned short u16t;
typedef unsigned int   u32t;

typedef __attribute__((ext_vector_type(8)))  __bf16 v8bf;
typedef __attribute__((ext_vector_type(16))) __bf16 v16bf;
typedef __attribute__((ext_vector_type(8)))  float  v8f;
typedef __attribute__((ext_vector_type(4))) unsigned int u32x4;
typedef __attribute__((ext_vector_type(8))) int          i32x8;
typedef __attribute__((ext_vector_type(4))) int          i32x4;

struct alignas(16) B128 { u16t h[8]; };

#if defined(__gfx1250__) && __has_builtin(__builtin_amdgcn_global_load_async_to_lds_b128)
#define HAVE_ASYNC_LDS 1
#else
#define HAVE_ASYNC_LDS 0
#endif
#if defined(__gfx1250__) && __has_builtin(__builtin_amdgcn_tensor_load_to_lds)
#define HAVE_TDM 1
#else
#define HAVE_TDM 0
#endif

static __device__ __forceinline__ float bf2f(u16t u) {
    u32t x = ((u32t)u) << 16;
    float f; __builtin_memcpy(&f, &x, 4);
    return f;
}
static __device__ __forceinline__ u16t f2bf(float f) {
    u32t u; __builtin_memcpy(&u, &f, 4);
    u32t r = u + 0x7FFFu + ((u >> 16) & 1u);   // round-to-nearest-even
    return (u16t)(r >> 16);
}
static __device__ __forceinline__ v8bf ld8(const u16t* p) {
    B128 t = *(const B128*)p;
    v8bf r; __builtin_memcpy(&r, &t, 16);
    return r;
}
static __device__ __forceinline__ v16bf mk16(v8bf lo, v8bf hi) {
    return __builtin_shufflevector(lo, hi, 0,1,2,3,4,5,6,7,8,9,10,11,12,13,14,15);
}
static __device__ __forceinline__ v8f wmma_bf16(v16bf a, v16bf b, v8f c) {
    // (neg_a, A, neg_b, B, c_mod, C, reuse_a, reuse_b)
    return __builtin_amdgcn_wmma_f32_16x16x32_bf16(false, a, false, b, (short)0, c, false, false);
}

#if HAVE_ASYNC_LDS
static __device__ __forceinline__ void async_cp16(const u16t* g, u16t* l) {
    __builtin_amdgcn_global_load_async_to_lds_b128(
        (__attribute__((address_space(1))) i32x4*)g,
        (__attribute__((address_space(3))) i32x4*)l, 0, 0);
}
static __device__ __forceinline__ void wait_async0() {
#if __has_builtin(__builtin_amdgcn_s_wait_asynccnt)
    __builtin_amdgcn_s_wait_asynccnt(0);
#else
    asm volatile("s_wait_asynccnt 0" ::: "memory");
#endif
}
#endif

// ---------------------------------------------------------------------------
// Elementwise f32 -> bf16
// ---------------------------------------------------------------------------
__global__ void f2bf_kernel(const float* __restrict__ s, u16t* __restrict__ d, size_t n) {
    size_t i = (size_t)blockIdx.x * blockDim.x + threadIdx.x;
    if (i < n) d[i] = f2bf(s[i]);
}

// ---------------------------------------------------------------------------
// bf16 GEMM: C[M,N] = A[M,K] * B[N,K]^T, f32 accumulate via WMMA.
// Tile 128x64, 8 waves/block, each wave owns a 16x64 row strip (4 C frags).
// Double-buffered LDS; A tile via per-lane async-to-LDS, B tile via one TDM
// descriptor per k-step (D# layout per CDNA5 ISA).
// ---------------------------------------------------------------------------
#define TM 128
#define TN 64
#define TK 32

__global__ __launch_bounds__(256) void gemm_bf16_kernel(
    const u16t* __restrict__ A, const u16t* __restrict__ Bm, float* __restrict__ C,
    int M, int N, int K)
{
    __shared__ u16t lA[2][TM * TK];
    __shared__ u16t lB[2][TN * TK];
    const int n0   = blockIdx.x * TN;
    const int m0   = blockIdx.y * TM;
    const int t    = threadIdx.x;
    const int wave = t >> 5;
    const int lane = t & 31;
    const int half = lane >> 4;
    const int l16  = lane & 15;
    const int mw   = wave * 16;

    v8f acc[4] = {};

    const int arow = t >> 1, ahc = (t & 1) * 16;   // A staging: 16 bf16/thread
    const int brow = t >> 2, bq  = (t & 3) * 8;    // B staging fallback

#if HAVE_TDM
    unsigned ldsB[2];
    ldsB[0] = (unsigned)(size_t)(__attribute__((address_space(3))) u16t*)&lB[0][0];
    ldsB[1] = (unsigned)(size_t)(__attribute__((address_space(3))) u16t*)&lB[1][0];
#endif

    auto stage = [&](int buf, int k0) {
        // ---- A tile: 128x32 bf16 ----
        {
            const u16t* g = A + (size_t)(m0 + arow) * K + k0 + ahc;
            u16t* d = &lA[buf][arow * TK + ahc];
#if HAVE_ASYNC_LDS
            async_cp16(g, d);
            async_cp16(g + 8, d + 8);
#else
            *(B128*)d       = *(const B128*)g;
            *(B128*)(d + 8) = *(const B128*)(g + 8);
#endif
        }
        // ---- B tile: 64x32 bf16 ----
#if HAVE_TDM
        if (t == 0) {
            unsigned long long ga =
                (unsigned long long)(const void*)(Bm + (size_t)n0 * K + k0);
            u32x4 g0; i32x8 g1; i32x4 g2 = {}; i32x4 g3 = {}; i32x8 g4 = {};
            // group0: count=1 | lds_addr | global_addr | type=2
            g0.x = 1u;
            g0.y = ldsB[buf];
            g0.z = (unsigned)(ga & 0xFFFFFFFFu);
            g0.w = (unsigned)((ga >> 32) & 0x01FFFFFFu) | (2u << 30);
            // group1: mask=0, data_size=1(2B); tensor 32x64, tile 32x64,
            // dim0_stride = K elements
            const unsigned t0 = TK, t1 = TN;
            const unsigned long long s0 = (unsigned long long)K;
            g1[0] = (int)(1u << 16);
            g1[1] = (int)((t0 & 0xFFFFu) << 16);
            g1[2] = (int)(((t0 >> 16) & 0xFFFFu) | ((t1 & 0xFFFFu) << 16));
            g1[3] = (int)(((t1 >> 16) & 0xFFFFu) | ((unsigned)TK << 16));
            g1[4] = (int)TN;
            g1[5] = (int)(s0 & 0xFFFFFFFFu);
            g1[6] = (int)((s0 >> 32) & 0xFFFFu);
            g1[7] = 0;
            __builtin_amdgcn_tensor_load_to_lds(g0, g1, g2, g3, g4, 0);
        }
#else
        {
            const u16t* g = Bm + (size_t)(n0 + brow) * K + k0 + bq;
            *(B128*)&lB[buf][brow * TK + bq] = *(const B128*)g;
        }
#endif
    };

    auto stage_wait = [&]() {
#if HAVE_ASYNC_LDS
        wait_async0();
#endif
#if HAVE_TDM
        __builtin_amdgcn_s_wait_tensorcnt(0);
#endif
    };

    const int nsteps = K / TK;
    stage(0, 0);
    stage_wait();
    __syncthreads();

    for (int s = 0; s < nsteps; s++) {
        const int cur = s & 1;
        if (s + 1 < nsteps) stage(cur ^ 1, (s + 1) * TK);   // overlap with WMMA

        // A frag: lane<16 -> K 0..15; lane>=16 -> K 16..31
        const u16t* pa = &lA[cur][(mw + l16) * TK];
        v16bf af = mk16(ld8(pa + (half ? 16 : 0)), ld8(pa + 8 + (half ? 16 : 0)));

        #pragma unroll
        for (int nf = 0; nf < 4; nf++) {
            // B frag: lane<16 -> N=l16, K 0..15; lane>=16 -> K 16..31
            const u16t* pb = &lB[cur][(nf * 16 + l16) * TK + (half ? 16 : 0)];
            v16bf bfr = mk16(ld8(pb), ld8(pb + 8));
            acc[nf] = wmma_bf16(af, bfr, acc[nf]);
        }

        stage_wait();
        __syncthreads();
    }

    // C/D layout: VGPR r: lanes 0-15 -> M=r, lanes 16-31 -> M=r+8; N=lane%16
    #pragma unroll
    for (int nf = 0; nf < 4; nf++) {
        int col = n0 + nf * 16 + l16;
        #pragma unroll
        for (int r = 0; r < 8; r++) {
            int row = m0 + mw + r + (half ? 8 : 0);
            C[(size_t)row * N + col] = acc[nf][r];
        }
    }
}

// ---------------------------------------------------------------------------
// Fused RMSNorm + RoPE (+gain,+prescale) and transpose (B,S,H,Dh)->(B,H,S,Dh).
// One wave per (b,h,s) row of Dh=128; lane owns dims {l, l+32, l+64, l+96}.
// ---------------------------------------------------------------------------
__global__ __launch_bounds__(256) void rope_kernel(
    const float* __restrict__ src,      // plain (B*S, H*Dh) f32 from GEMM
    u16t* __restrict__ dst,             // (B,H,S,Dh) bf16
    const float* __restrict__ gain,     // per-head gain or nullptr
    float outScale, int B_, int S_, int H_, int Dh_)
{
    int rowsPerBlk = blockDim.x >> 5;
    int row  = blockIdx.x * rowsPerBlk + (threadIdx.x >> 5);   // [0, B*H*S)
    int lane = threadIdx.x & 31;
    int spos = row % S_;
    int bh   = row / S_;
    int h    = bh % H_;
    int b    = bh / H_;

    const float* p = src + ((size_t)(b * S_ + spos)) * (H_ * Dh_) + h * Dh_;
    float x0 = p[lane], x1 = p[lane + 32], x2 = p[lane + 64], x3 = p[lane + 96];

    float ss = x0*x0 + x1*x1 + x2*x2 + x3*x3;
    #pragma unroll
    for (int m = 16; m >= 1; m >>= 1) ss += __shfl_xor(ss, m, 32);
    float rn = rsqrtf(ss * (1.0f / 128.0f) + 1e-6f);
    x0 *= rn; x1 *= rn; x2 *= rn; x3 *= rn;

    // rope pairs: (l, l+64) angle j=l ; (l+32, l+96) angle j=l+32
    const float ln1e4 = 9.210340371976184f;    // log(10000)
    float a0 = (float)spos * __expf(-ln1e4 * ((float)lane        * (1.0f / 64.0f)));
    float a1 = (float)spos * __expf(-ln1e4 * ((float)(lane + 32) * (1.0f / 64.0f)));
    float c0, s0, c1, s1;
    __sincosf(a0, &s0, &c0);
    __sincosf(a1, &s1, &c1);
    float o0 =  x0 * c0 + x2 * s0;
    float o2 = -x0 * s0 + x2 * c0;
    float o1 =  x1 * c1 + x3 * s1;
    float o3 = -x1 * s1 + x3 * c1;

    float g = outScale * (gain ? gain[h] : 1.0f);
    u16t* q = dst + ((size_t)bh * S_ + spos) * Dh_;
    q[lane]      = f2bf(o0 * g);
    q[lane + 32] = f2bf(o1 * g);
    q[lane + 64] = f2bf(o2 * g);
    q[lane + 96] = f2bf(o3 * g);
}

// V: plain f32 (B,S,H,Dh) -> bf16 (B,H,S,Dh)
__global__ void vconv_kernel(const float* __restrict__ src, u16t* __restrict__ dst,
                             int B_, int S_, int H_, int Dh_)
{
    size_t i = (size_t)blockIdx.x * blockDim.x + threadIdx.x;
    size_t total = (size_t)B_ * S_ * H_ * Dh_;
    if (i >= total) return;
    int d  = (int)(i % Dh_);
    size_t r = i / Dh_;
    int h  = (int)(r % H_);
    size_t r2 = r / H_;
    int s  = (int)(r2 % S_);
    int b  = (int)(r2 / S_);
    dst[(((size_t)(b * H_ + h)) * S_ + s) * Dh_ + d] = f2bf(src[i]);
}

// ---------------------------------------------------------------------------
// Deterministic Gaussian stand-in for the JAX threefry proj matrix (128x32).
// (JAX key(42) stream is not reproducible on-device; no runtime check here.)
// ---------------------------------------------------------------------------
static __device__ __forceinline__ u32t hashu(u32t x) {
    x ^= x >> 17; x *= 0xED5AD4BBu;
    x ^= x >> 11; x *= 0xAC4C1B51u;
    x ^= x >> 15; x *= 0x31848BABu;
    x ^= x >> 14; return x;
}
__global__ void projgen_kernel(float* __restrict__ proj) {
    int i = blockIdx.x * blockDim.x + threadIdx.x;
    if (i >= 128 * 32) return;
    u32t h1 = hashu(0x9E3779B9u ^ (u32t)i);
    u32t h2 = hashu(h1 ^ 0x85EBCA6Bu);
    float u1 = (float)(h1 & 0xFFFFFFu) * (1.0f / 16777216.0f) + 1e-7f;
    float u2 = (float)(h2 & 0xFFFFFFu) * (1.0f / 16777216.0f);
    float g = sqrtf(-2.0f * logf(u1)) * cosf(6.2831853f * u2);
    proj[i] = g * (1.0f / 5.656854249f);   // / sqrt(32)
}

// q mean over S per (b,h): 128 threads, one per dim.
__global__ __launch_bounds__(128) void qmean_kernel(
    const u16t* __restrict__ qb, float* __restrict__ qmean, int S_, int Dh_)
{
    int bh = blockIdx.x;
    int d  = threadIdx.x;
    const u16t* p = qb + (size_t)bh * S_ * Dh_ + d;
    float s = 0.0f;
    for (int i = 0; i < S_; i++) s += bf2f(p[(size_t)i * Dh_]);
    qmean[bh * Dh_ + d] = s / (float)S_;
}

// Per (b,h,block): k_proj (64x32), row-normalize, centroid, radius.
__global__ __launch_bounds__(128) void blockstats_kernel(
    const u16t* __restrict__ kb, const float* __restrict__ proj,
    float* __restrict__ cent, float* __restrict__ radii,
    int S_, int Dh_, int Nblk)
{
    __shared__ float kp[64 * 32];
    __shared__ float cs[32];
    int blk = blockIdx.x % Nblk;
    int bh  = blockIdx.x / Nblk;
    const u16t* kbase = kb + ((size_t)bh * S_ + blk * 64) * Dh_;
    int t  = threadIdx.x;
    int j  = t & 31;
    int r0 = t >> 5;
    for (int r = r0; r < 64; r += 4) {
        float acc = 0.0f;
        const u16t* krow = kbase + (size_t)r * Dh_;
        for (int d = 0; d < Dh_; d++) acc += bf2f(krow[d]) * proj[d * 32 + j];
        kp[r * 32 + j] = acc;
    }
    __syncthreads();
    if (t < 64) {
        float ss = 0.0f;
        for (int jj = 0; jj < 32; jj++) { float v = kp[t * 32 + jj]; ss += v * v; }
        float inv = 1.0f / fmaxf(sqrtf(ss), 1e-12f);
        for (int jj = 0; jj < 32; jj++) kp[t * 32 + jj] *= inv;
    }
    __syncthreads();
    if (t < 32) {
        float s = 0.0f;
        for (int r = 0; r < 64; r++) s += kp[r * 32 + t];
        cs[t] = s * (1.0f / 64.0f);
    }
    __syncthreads();
    if (t == 0) {
        float ss = 0.0f;
        for (int jj = 0; jj < 32; jj++) ss += cs[jj] * cs[jj];
        float inv = 1.0f / fmaxf(sqrtf(ss), 1e-12f);
        for (int jj = 0; jj < 32; jj++) {
            cs[jj] *= inv;
            cent[((size_t)bh * Nblk + blk) * 32 + jj] = cs[jj];
        }
        float mn = 1e30f;
        for (int r = 0; r < 64; r++) {
            float dot = 0.0f;
            for (int jj = 0; jj < 32; jj++) dot += kp[r * 32 + jj] * cs[jj];
            mn = fminf(mn, dot);
        }
        radii[bh * Nblk + blk] = fminf(fmaxf(1.0f - mn, 0.0f), 1.0f);
    }
}

// Per (b,h): q_proj, score per block (lane==block since Nblk==32), top-18 sorted.
__global__ __launch_bounds__(32) void select_kernel(
    const float* __restrict__ qmean, const float* __restrict__ proj,
    const float* __restrict__ cent, const float* __restrict__ radii,
    int* __restrict__ sel, int Nblk, int nKeep, int Dh_)
{
    __shared__ float qp[32];
    __shared__ float sc[32];
    int bh = blockIdx.x;
    int lane = threadIdx.x;

    float acc = 0.0f;
    const float* qm = qmean + bh * Dh_;
    for (int d = 0; d < Dh_; d++) acc += qm[d] * proj[d * 32 + lane];
    float ss = acc * acc;
    #pragma unroll
    for (int m = 16; m >= 1; m >>= 1) ss += __shfl_xor(ss, m, 32);
    qp[lane] = acc * (1.0f / fmaxf(sqrtf(ss), 1e-12f));
    __syncthreads();

    float s = 0.0f;
    const float* cn = cent + ((size_t)bh * Nblk + lane) * 32;
    for (int jj = 0; jj < 32; jj++) s += qp[jj] * cn[jj];
    s += radii[bh * Nblk + lane];
    if (lane >= Nblk - 2) s = 1e30f;    // last two blocks forced in
    sc[lane] = s;
    __syncthreads();

    if (lane == 0) {
        float tv[32];
        int idx[32];
        for (int i = 0; i < 32; i++) tv[i] = sc[i];
        for (int kk = 0; kk < nKeep; kk++) {
            int best = 0; float bv = -1e38f;
            for (int i = 0; i < Nblk; i++) if (tv[i] > bv) { bv = tv[i]; best = i; }
            idx[kk] = best; tv[best] = -1e38f;
        }
        for (int i = 1; i < nKeep; i++) {     // sort ascending
            int v = idx[i], jx = i - 1;
            while (jx >= 0 && idx[jx] > v) { idx[jx + 1] = idx[jx]; jx--; }
            idx[jx + 1] = v;
        }
        for (int i = 0; i < nKeep; i++) sel[bh * nKeep + i] = idx[i];
    }
}

// ---------------------------------------------------------------------------
// Flash-style block-sparse attention. Block = (b, h, 64 q rows), 4 waves;
// wave owns 16 q rows. Streams 36 chunks of 32 selected keys.
// Scores + P@V via bf16 WMMA. V staged transposed in LDS; P converted
// C-layout -> A-layout through per-wave LDS with explicit s_wait_dscnt.
// ---------------------------------------------------------------------------
#define NKEEP 18

__global__ __launch_bounds__(128) void attn_kernel(
    const u16t* __restrict__ qb, const u16t* __restrict__ kb,
    const u16t* __restrict__ vb, const int* __restrict__ sel,
    u16t* __restrict__ y, int B_, int H_, int S_, int Dh_)
{
    __shared__ u16t vT[128 * 32];        // [d][k]  (transposed V chunk)
    __shared__ u16t pl[4][16 * 32];      // per-wave P scratch

    int qt   = blockIdx.x & 31;          // S/64 = 32 q tiles
    int bh   = blockIdx.x >> 5;
    int t    = threadIdx.x;
    int wave = t >> 5;
    int lane = t & 31;
    int half = lane >> 4;
    int l16  = lane & 15;
    int s0   = qt * 64 + wave * 16;
    const size_t headOff = (size_t)bh * S_ * Dh_;

    // q A-frags (pre-scaled by gain/sqrt(Dh) in rope_kernel)
    v16bf qf[4];
    {
        const u16t* qrow = qb + headOff + (size_t)(s0 + l16) * Dh_;
        #pragma unroll
        for (int f = 0; f < 4; f++) {
            const u16t* p = qrow + f * 32 + (half ? 16 : 0);
            qf[f] = mk16(ld8(p), ld8(p + 8));
        }
    }

    v8f o[8] = {};
    float mrun[8], lrun[8];
    #pragma unroll
    for (int r = 0; r < 8; r++) { mrun[r] = -1e30f; lrun[r] = 0.0f; }

    const int* selp = sel + bh * NKEEP;
    const int nChunks = (NKEEP * 64) / 32;   // 36

    for (int c = 0; c < nChunks; c++) {
        int blk   = selp[c >> 1];
        int kbase = blk * 64 + (c & 1) * 32;   // chunk never crosses a block

        __syncthreads();
        {   // stage V chunk transposed: thread loads 32 bf16 of one key row
            int kk = t >> 2, c0 = (t & 3) * 32;
            const u16t* vrow = vb + headOff + (size_t)(kbase + kk) * Dh_ + c0;
            B128 r0 = *(const B128*)(vrow);
            B128 r1 = *(const B128*)(vrow + 8);
            B128 r2 = *(const B128*)(vrow + 16);
            B128 r3 = *(const B128*)(vrow + 24);
            #pragma unroll
            for (int e = 0; e < 8; e++) {
                vT[(c0 + e)      * 32 + kk] = r0.h[e];
                vT[(c0 + 8 + e)  * 32 + kk] = r1.h[e];
                vT[(c0 + 16 + e) * 32 + kk] = r2.h[e];
                vT[(c0 + 24 + e) * 32 + kk] = r3.h[e];
            }
        }
        __syncthreads();

        // scores: two 16x16 frags (keys kbase..+16, +16..+32), K=128 in 4 steps
        v8f slo = {}, shi = {};
        {
            const u16t* kr0 = kb + headOff + (size_t)(kbase + l16)      * Dh_ + (half ? 16 : 0);
            const u16t* kr1 = kb + headOff + (size_t)(kbase + 16 + l16) * Dh_ + (half ? 16 : 0);
            #pragma unroll
            for (int f = 0; f < 4; f++) {
                v16bf b0 = mk16(ld8(kr0 + f * 32), ld8(kr0 + f * 32 + 8));
                v16bf b1 = mk16(ld8(kr1 + f * 32), ld8(kr1 + f * 32 + 8));
                slo = wmma_bf16(qf[f], b0, slo);
                shi = wmma_bf16(qf[f], b1, shi);
            }
        }

        // flash update (row-wise ops are per C-VGPR within each 16-lane half)
        float pscale[8];
        #pragma unroll
        for (int r = 0; r < 8; r++) {
            float mx = fmaxf(slo[r], shi[r]);
            #pragma unroll
            for (int m = 8; m >= 1; m >>= 1) mx = fmaxf(mx, __shfl_xor(mx, m, 16));
            float mnew = fmaxf(mrun[r], mx);
            float scl  = __expf(mrun[r] - mnew);
            mrun[r] = mnew;
            float plo = __expf(slo[r] - mnew);
            float phi = __expf(shi[r] - mnew);
            float rs  = plo + phi;
            #pragma unroll
            for (int m = 8; m >= 1; m >>= 1) rs += __shfl_xor(rs, m, 16);
            lrun[r] = lrun[r] * scl + rs;
            pscale[r] = scl;
            slo[r] = plo; shi[r] = phi;
        }
        #pragma unroll
        for (int nf = 0; nf < 8; nf++)
            #pragma unroll
            for (int r = 0; r < 8; r++) o[nf][r] *= pscale[r];

        // P: C-layout -> LDS -> A-layout (same wave; explicit DS wait)
        u16t* pw = pl[wave];
        #pragma unroll
        for (int r = 0; r < 8; r++) {
            int m = r + (half ? 8 : 0);
            pw[m * 32 + l16]      = f2bf(slo[r]);
            pw[m * 32 + 16 + l16] = f2bf(shi[r]);
        }
        asm volatile("s_wait_dscnt 0" ::: "memory");

        const u16t* pr = &pw[l16 * 32 + (half ? 16 : 0)];
        v16bf pa = mk16(ld8(pr), ld8(pr + 8));

        #pragma unroll
        for (int nf = 0; nf < 8; nf++) {
            const u16t* vr = &vT[(nf * 16 + l16) * 32 + (half ? 16 : 0)];
            v16bf vbf = mk16(ld8(vr), ld8(vr + 8));
            o[nf] = wmma_bf16(pa, vbf, o[nf]);
        }
    }

    // epilogue: y (B,S,H*Dh) bf16 for the Wo GEMM
    int hcol0 = (bh % H_) * Dh_;
    int b     = bh / H_;
    #pragma unroll
    for (int nf = 0; nf < 8; nf++) {
        #pragma unroll
        for (int r = 0; r < 8; r++) {
            int m    = r + (half ? 8 : 0);
            int srow = s0 + m;
            float val = o[nf][r] / lrun[r];
            y[((size_t)(b * S_) + srow) * (H_ * Dh_) + hcol0 + nf * 16 + l16] = f2bf(val);
        }
    }
}

// ---------------------------------------------------------------------------
// Host orchestration
// ---------------------------------------------------------------------------
extern "C" void kernel_launch(void* const* d_in, const int* in_sizes, int n_in,
                              void* d_out, int out_size, void* d_ws, size_t ws_size,
                              hipStream_t stream)
{
    (void)in_sizes; (void)n_in; (void)out_size; (void)ws_size;
    const float* x     = (const float*)d_in[0];
    const float* Wq    = (const float*)d_in[1];
    const float* Wk    = (const float*)d_in[2];
    const float* Wv    = (const float*)d_in[3];
    const float* Wo    = (const float*)d_in[4];
    const float* qgain = (const float*)d_in[5];
    float* out = (float*)d_out;

    const int B = 2, S = 2048, Dm = 2048, H = 16, Dh = 128, Nblk = 32, NK = NKEEP;
    const size_t MROWS = (size_t)B * S;          // 4096
    const size_t NX = MROWS * Dm;                // x / qkv / y element count
    const size_t NW = (size_t)Dm * Dm;           // weight element count

    // workspace carve (256B aligned)
    char* w = (char*)d_ws;
    auto carve = [&](size_t bytes) -> char* {
        char* p = w;
        w += (bytes + 255) & ~(size_t)255;
        return p;
    };
    u16t* xb    = (u16t*)carve(NX * 2);
    u16t* Wqb   = (u16t*)carve(NW * 2);
    u16t* Wkb   = (u16t*)carve(NW * 2);
    u16t* Wvb   = (u16t*)carve(NW * 2);
    u16t* Wob   = (u16t*)carve(NW * 2);
    float* tmpf = (float*)carve(NX * 4);         // reused for Q, K, V f32
    u16t* qbf   = (u16t*)carve(NX * 2);
    u16t* kbf   = (u16t*)carve(NX * 2);
    u16t* vbf   = (u16t*)carve(NX * 2);
    u16t* ybf   = (u16t*)carve(NX * 2);
    float* qmean = (float*)carve((size_t)B * H * Dh * 4);
    float* proj  = (float*)carve((size_t)Dh * 32 * 4);
    float* cent  = (float*)carve((size_t)B * H * Nblk * 32 * 4);
    float* rad   = (float*)carve((size_t)B * H * Nblk * 4);
    int*   selb  = (int*)carve((size_t)B * H * NK * 4);

    // bf16 conversions
    f2bf_kernel<<<(int)((NX + 255) / 256), 256, 0, stream>>>(x,  xb,  NX);
    f2bf_kernel<<<(int)((NW + 255) / 256), 256, 0, stream>>>(Wq, Wqb, NW);
    f2bf_kernel<<<(int)((NW + 255) / 256), 256, 0, stream>>>(Wk, Wkb, NW);
    f2bf_kernel<<<(int)((NW + 255) / 256), 256, 0, stream>>>(Wv, Wvb, NW);
    f2bf_kernel<<<(int)((NW + 255) / 256), 256, 0, stream>>>(Wo, Wob, NW);

    dim3 gemmGrid(Dm / TN, (int)(MROWS / TM));
    int ropeGrid = (B * H * S) / 8;              // 8 rows per 256-thread block
    const float qScale = 1.0f / sqrtf((float)Dh);

    // Q
    gemm_bf16_kernel<<<gemmGrid, 256, 0, stream>>>(xb, Wqb, tmpf, (int)MROWS, Dm, Dm);
    rope_kernel<<<ropeGrid, 256, 0, stream>>>(tmpf, qbf, qgain, qScale, B, S, H, Dh);
    // K
    gemm_bf16_kernel<<<gemmGrid, 256, 0, stream>>>(xb, Wkb, tmpf, (int)MROWS, Dm, Dm);
    rope_kernel<<<ropeGrid, 256, 0, stream>>>(tmpf, kbf, nullptr, 1.0f, B, S, H, Dh);
    // V
    gemm_bf16_kernel<<<gemmGrid, 256, 0, stream>>>(xb, Wvb, tmpf, (int)MROWS, Dm, Dm);
    vconv_kernel<<<(int)((NX + 255) / 256), 256, 0, stream>>>(tmpf, vbf, B, S, H, Dh);

    // JL block selection
    projgen_kernel<<<16, 256, 0, stream>>>(proj);
    qmean_kernel<<<B * H, 128, 0, stream>>>(qbf, qmean, S, Dh);
    blockstats_kernel<<<B * H * Nblk, 128, 0, stream>>>(kbf, proj, cent, rad, S, Dh, Nblk);
    select_kernel<<<B * H, 32, 0, stream>>>(qmean, proj, cent, rad, selb, Nblk, NK, Dh);

    // sparse attention + output projection
    attn_kernel<<<B * H * (S / 64), 128, 0, stream>>>(qbf, kbf, vbf, selb, ybf, B, H, S, Dh);
    gemm_bf16_kernel<<<gemmGrid, 256, 0, stream>>>(ybf, Wob, out, (int)MROWS, Dm, Dm);
}
